// HierPRBlock_76312978916154
// MI455X (gfx1250) — matrix operand order repaired and checked by reading, not compiled, not run on hardware
//
#include <hip/hip_runtime.h>

typedef __attribute__((ext_vector_type(16))) _Float16 v16h;
typedef __attribute__((ext_vector_type(8)))  float    v8f;

#define BB 8
#define HH 512
#define WW 512
#define HWSZ (HH*WW)               // 262144
#define KSEL (HWSZ/10)             // 26214
#define CF 256
#define HF 128
#define WF 128
#define FSTRIDE (HF*WF)            // 16384
#define NPTS 64
#define NTILES_PT ((KSEL + NPTS - 1)/NPTS)  // 410
#define KB1 9                      // layer1 k-blocks: 257 padded to 288
#define KB2 8                      // layer2 k-blocks: 256
#define EPSBN 1e-5f

// ---- workspace layout (bytes) ----
#define WS_HIST   0u               // 8*256*4 = 8192
#define WS_PREFIX 8192u            // 8*4
#define WS_BASE   8224u            // 8*4
#define WS_LESS   8256u            // 8*4
#define WS_EQ     8288u            // 8*4
#define WS_SEL    8448u            // 8*26214*4 = 838848 -> ends 847296
#define WS_W1P    847360u          // 16*9*32*16 f16 = 147456 -> ends 994816
#define WS_W2P    994816u          // 16*8*32*16 f16 = 131072 -> ends 1125888
#define WS_B1F    1125888u         // 256 f32
#define WS_B2F    1126912u         // 256 f32

__device__ __forceinline__ int kfun(int g, int s) {
  // contiguous WMMA 16-bit A/B k order: k = 16*g + s, slot s (0..15), lane-group g (0/1)
  return s + 16 * g;
}
__device__ __forceinline__ unsigned ordkey(float f) {
  unsigned u = __float_as_uint(f);
  return (u & 0x80000000u) ? ~u : (u | 0x80000000u);  // ascending-orderable
}

// ---------------- out = prev_mask ----------------
__global__ void copy_mask_kernel(const float4* __restrict__ src, float4* __restrict__ dst) {
  int i = blockIdx.x * 256 + threadIdx.x;
  dst[i] = src[i];
}

// ---------------- radix select (K smallest per batch) ----------------
__global__ void sel_init_kernel(char* ws) {
  unsigned* hist = (unsigned*)(ws + WS_HIST);
  int t = threadIdx.x;
  for (int i = t; i < 2048; i += 256) hist[i] = 0u;
  if (t < 8) {
    ((unsigned*)(ws + WS_PREFIX))[t] = 0u;
    ((unsigned*)(ws + WS_BASE))[t]   = 0u;
    ((unsigned*)(ws + WS_LESS))[t]   = 0u;
    ((unsigned*)(ws + WS_EQ))[t]     = 0u;
  }
}

__global__ void sel_hist_kernel(const float* __restrict__ unc, char* ws, int r) {
  unsigned* hist = (unsigned*)(ws + WS_HIST);
  const unsigned* prefix = (const unsigned*)(ws + WS_PREFIX);
  int b = blockIdx.x >> 5;
  int chunk = blockIdx.x & 31;
  unsigned pfx = prefix[b];
  int base = b * HWSZ + chunk * 8192 + threadIdx.x;
  for (int k = 0; k < 32; k++) {
    unsigned key = ordkey(unc[base + k * 256]);
    bool match = (r == 0) || ((key >> (32 - 8 * r)) == pfx);
    if (match) atomicAdd(&hist[b * 256 + ((key >> (24 - 8 * r)) & 255u)], 1u);
  }
}

__global__ void sel_scan_kernel(char* ws) {
  unsigned* hist   = (unsigned*)(ws + WS_HIST);
  unsigned* prefix = (unsigned*)(ws + WS_PREFIX);
  unsigned* basep  = (unsigned*)(ws + WS_BASE);
  int t = threadIdx.x;
  if (t < 8) {
    unsigned acc = basep[t];
    unsigned pfx = prefix[t];
    int v, sv = 255;
    for (v = 0; v < 256; v++) {
      unsigned h = hist[t * 256 + v];
      if (acc + h >= (unsigned)KSEL) { sv = v; break; }
      acc += h;
    }
    prefix[t] = (pfx << 8) | (unsigned)sv;
    basep[t]  = acc;
  }
  __syncthreads();
  for (int i = t; i < 2048; i += 256) hist[i] = 0u;  // ready for next round
}

__global__ void sel_compact_kernel(const float* __restrict__ unc, char* ws) {
  const unsigned* prefix = (const unsigned*)(ws + WS_PREFIX);
  const unsigned* basep  = (const unsigned*)(ws + WS_BASE);
  unsigned* lessc = (unsigned*)(ws + WS_LESS);
  unsigned* eqc   = (unsigned*)(ws + WS_EQ);
  int* sel = (int*)(ws + WS_SEL);
  int b = blockIdx.x >> 5;
  int chunk = blockIdx.x & 31;
  unsigned thr  = prefix[b];
  unsigned base = basep[b];
  unsigned E = (unsigned)KSEL - base;
  int off = chunk * 8192 + threadIdx.x;
  for (int k = 0; k < 32; k++) {
    int i = off + k * 256;
    unsigned key = ordkey(unc[b * HWSZ + i]);
    if (key < thr) {
      unsigned p = atomicAdd(&lessc[b], 1u);
      sel[b * KSEL + p] = i;
    } else if (key == thr) {
      unsigned e = atomicAdd(&eqc[b], 1u);
      if (e < E) sel[b * KSEL + base + e] = i;
    }
  }
}

// ---------------- BN-fold + f16 pre-swizzle weights into WMMA A layout ----------------
__global__ void prep_weights_kernel(const float* __restrict__ w1, const float* __restrict__ b1,
                                    const float* __restrict__ g1, const float* __restrict__ be1,
                                    const float* __restrict__ m1, const float* __restrict__ v1,
                                    const float* __restrict__ w2, const float* __restrict__ b2,
                                    const float* __restrict__ g2, const float* __restrict__ be2,
                                    const float* __restrict__ m2, const float* __restrict__ v2,
                                    char* ws) {
  _Float16* w1p = (_Float16*)(ws + WS_W1P);
  _Float16* w2p = (_Float16*)(ws + WS_W2P);
  float* b1f = (float*)(ws + WS_B1F);
  float* b2f = (float*)(ws + WS_B2F);
  int id = blockIdx.x * 256 + threadIdx.x;
  const int N1 = 16 * KB1 * 32 * 16;  // 73728
  const int N2 = 16 * KB2 * 32 * 16;  // 65536
  if (id < N1) {
    int slot = id & 15, lane = (id >> 4) & 31, kb = (id >> 9) % KB1, mt = (id >> 9) / KB1;
    int k = kb * 32 + kfun(lane >> 4, slot);
    int m = mt * 16 + (lane & 15);
    float s = g1[m] * rsqrtf(v1[m] + EPSBN);
    float val = (k < 257) ? s * w1[m * 257 + k] : 0.0f;  // k=0 coarse, k=1..256 fine
    w1p[id] = (_Float16)val;
  } else if (id < N1 + N2) {
    int j = id - N1;
    int slot = j & 15, lane = (j >> 4) & 31, kb = (j >> 9) % KB2, mt = (j >> 9) / KB2;
    int k = kb * 32 + kfun(lane >> 4, slot);
    int m = mt * 16 + (lane & 15);
    float s = g2[m] * rsqrtf(v2[m] + EPSBN);
    w2p[j] = (_Float16)(s * w2[m * 256 + k]);
  } else if (id < N1 + N2 + 256) {
    int m = id - N1 - N2;
    float s = g1[m] * rsqrtf(v1[m] + EPSBN);
    b1f[m] = s * (b1[m] - m1[m]) + be1[m];
  } else if (id < N1 + N2 + 512) {
    int m = id - N1 - N2 - 256;
    float s = g2[m] * rsqrtf(v2[m] + EPSBN);
    b2f[m] = s * (b2[m] - m2[m]) + be2[m];
  }
}

// ---------------- main: gather + WMMA MLP + scatter ----------------
__global__ __launch_bounds__(256) void hier_main_kernel(
    const float* __restrict__ prev_mask, const float* __restrict__ enc_f,
    const float* __restrict__ w3g, const float* __restrict__ b3g,
    const char* __restrict__ wsc, float* __restrict__ out) {
  const _Float16* w1p = (const _Float16*)(wsc + WS_W1P);
  const _Float16* w2p = (const _Float16*)(wsc + WS_W2P);
  const float* b1f = (const float*)(wsc + WS_B1F);
  const float* b2f = (const float*)(wsc + WS_B2F);
  const int* sel = (const int*)(wsc + WS_SEL);

  __shared__ __align__(32) _Float16 A0[4 * KB1 * 32 * 16];  // 36 KB, B-operand layout
  __shared__ __align__(32) _Float16 A1[4 * KB2 * 32 * 16];  // 32 KB
  __shared__ int   gofs[NPTS][4];
  __shared__ float wgt[NPTS][4];
  __shared__ int   pidx[NPTS];
  __shared__ int   pval[NPTS];
  __shared__ float w3s[256];

  int tid  = threadIdx.x;
  int b    = blockIdx.x / NTILES_PT;
  int tile = blockIdx.x % NTILES_PT;

  w3s[tid] = w3g[tid];
  {  // zero A0 (covers K-pad 257..287 and tail points)
    float4 z = make_float4(0.f, 0.f, 0.f, 0.f);
    float4* a0v = (float4*)A0;
    for (int i = tid; i < (4 * KB1 * 32 * 16) / 8; i += 256) a0v[i] = z;
  }
  __syncthreads();

  // ---- point metadata + coarse channel (k=0) ----
  if (tid < NPTS) {
    int slot = tile * NPTS + tid;
    int valid = slot < KSEL;
    int idx = valid ? sel[b * KSEL + slot] : 0;
    pidx[tid] = idx; pval[tid] = valid;
    int col = idx & (WW - 1);
    int row = idx >> 9;
    // grid_sample coords on 128x128: x = (col+0.5)/4 - 0.5
    float x = ((float)col + 0.5f) * 0.25f - 0.5f;
    float y = ((float)row + 0.5f) * 0.25f - 0.5f;
    float x0f = floorf(x), y0f = floorf(y);
    float wx = x - x0f, wy = y - y0f;
    int x0 = (int)x0f, y0 = (int)y0f;
    float cw[4] = { (1.f - wx) * (1.f - wy), wx * (1.f - wy), (1.f - wx) * wy, wx * wy };
    int cx[4] = { x0, x0 + 1, x0, x0 + 1 };
    int cy[4] = { y0, y0, y0 + 1, y0 + 1 };
    for (int c = 0; c < 4; c++) {
      bool ok = (cx[c] >= 0) && (cx[c] < WF) && (cy[c] >= 0) && (cy[c] < HF);
      int xx = min(max(cx[c], 0), WF - 1), yy = min(max(cy[c], 0), HF - 1);
      wgt[tid][c]  = ok ? cw[c] : 0.0f;
      gofs[tid][c] = b * CF * FSTRIDE + yy * WF + xx;
    }
    float coarse = prev_mask[b * HWSZ + idx];
    int nt = tid >> 4, nl = tid & 15;
    A0[((nt * KB1 + 0) * 32 + nl) * 16 + 0] = (_Float16)coarse;  // k=0 -> lane=nl, slot=0
  }
  __syncthreads();

  // ---- bilinear gather of 256 fine channels into B-operand layout ----
  {
    int n = tid & 63;
    int cgrp = tid >> 6;  // 4 thread-groups x 64 channels
    int o0 = gofs[n][0], o1 = gofs[n][1], o2 = gofs[n][2], o3 = gofs[n][3];
    float q0 = wgt[n][0], q1 = wgt[n][1], q2 = wgt[n][2], q3 = wgt[n][3];
    int nt = n >> 4, nl = n & 15;
    for (int cc = 0; cc < 64; cc++) {
      int c = cgrp * 64 + cc;
      int co = c * FSTRIDE;
      float val = q0 * enc_f[o0 + co] + q1 * enc_f[o1 + co] +
                  q2 * enc_f[o2 + co] + q3 * enc_f[o3 + co];
      int k = c + 1;
      int kb = k >> 5, ki = k & 31;
      int lane = nl + ((ki >= 16) ? 16 : 0);
      int slot = ki & 15;
      A0[((nt * KB1 + kb) * 32 + lane) * 16 + slot] = (_Float16)val;
    }
  }
  __syncthreads();

  int wave = tid >> 5;
  int lane = tid & 31;
  int lg = lane >> 4;
  int nl = lane & 15;

  // ---- layer 1: [256x288] x [288x64], WMMA f16 -> f32 ----
  for (int mi = 0; mi < 2; mi++) {
    int mt = wave * 2 + mi;
    float4 bA = *(const float4*)(b1f + mt * 16 + 8 * lg);
    float4 bB = *(const float4*)(b1f + mt * 16 + 8 * lg + 4);
    float bias[8] = { bA.x, bA.y, bA.z, bA.w, bB.x, bB.y, bB.z, bB.w };
    v8f acc0 = {0,0,0,0,0,0,0,0}, acc1 = {0,0,0,0,0,0,0,0};
    v8f acc2 = {0,0,0,0,0,0,0,0}, acc3 = {0,0,0,0,0,0,0,0};
    for (int kb = 0; kb < KB1; kb++) {
      v16h a = *(const v16h*)(w1p + ((mt * KB1 + kb) * 32 + lane) * 16);
      v16h bm0 = *(const v16h*)(&A0[((0 * KB1 + kb) * 32 + lane) * 16]);
      v16h bm1 = *(const v16h*)(&A0[((1 * KB1 + kb) * 32 + lane) * 16]);
      v16h bm2 = *(const v16h*)(&A0[((2 * KB1 + kb) * 32 + lane) * 16]);
      v16h bm3 = *(const v16h*)(&A0[((3 * KB1 + kb) * 32 + lane) * 16]);
      acc0 = __builtin_amdgcn_wmma_f32_16x16x32_f16(false, a, false, bm0, (short)0, acc0, false, false);
      acc1 = __builtin_amdgcn_wmma_f32_16x16x32_f16(false, a, false, bm1, (short)0, acc1, false, false);
      acc2 = __builtin_amdgcn_wmma_f32_16x16x32_f16(false, a, false, bm2, (short)0, acc2, false, false);
      acc3 = __builtin_amdgcn_wmma_f32_16x16x32_f16(false, a, false, bm3, (short)0, acc3, false, false);
    }
    // bias + ReLU, repack 8 contiguous k as f16 -> single 16B LDS store (B layout for layer 2)
    int kbase = mt * 16 + 8 * lg;
    int kb2 = kbase >> 5;
    int lsel = ((kbase & 31) >= 16) ? 16 : 0;
    int lslot = kbase & 15;  // 0 or 8
    v8f accs[4] = { acc0, acc1, acc2, acc3 };
    for (int nt = 0; nt < 4; nt++) {
      __align__(16) _Float16 hv[8];
#pragma unroll
      for (int v = 0; v < 8; v++) {
        float r = accs[nt][v] + bias[v];
        hv[v] = (_Float16)fmaxf(r, 0.0f);
      }
      *(float4*)(&A1[((nt * KB2 + kb2) * 32 + nl + lsel) * 16 + lslot]) = *(const float4*)hv;
    }
  }
  __syncthreads();

  // ---- layer 2: [256x256] x [256x64] ----
  for (int mi = 0; mi < 2; mi++) {
    int mt = wave * 2 + mi;
    float4 bA = *(const float4*)(b2f + mt * 16 + 8 * lg);
    float4 bB = *(const float4*)(b2f + mt * 16 + 8 * lg + 4);
    float bias[8] = { bA.x, bA.y, bA.z, bA.w, bB.x, bB.y, bB.z, bB.w };
    v8f acc0 = {0,0,0,0,0,0,0,0}, acc1 = {0,0,0,0,0,0,0,0};
    v8f acc2 = {0,0,0,0,0,0,0,0}, acc3 = {0,0,0,0,0,0,0,0};
    for (int kb = 0; kb < KB2; kb++) {
      v16h a = *(const v16h*)(w2p + ((mt * KB2 + kb) * 32 + lane) * 16);
      v16h bm0 = *(const v16h*)(&A1[((0 * KB2 + kb) * 32 + lane) * 16]);
      v16h bm1 = *(const v16h*)(&A1[((1 * KB2 + kb) * 32 + lane) * 16]);
      v16h bm2 = *(const v16h*)(&A1[((2 * KB2 + kb) * 32 + lane) * 16]);
      v16h bm3 = *(const v16h*)(&A1[((3 * KB2 + kb) * 32 + lane) * 16]);
      acc0 = __builtin_amdgcn_wmma_f32_16x16x32_f16(false, a, false, bm0, (short)0, acc0, false, false);
      acc1 = __builtin_amdgcn_wmma_f32_16x16x32_f16(false, a, false, bm1, (short)0, acc1, false, false);
      acc2 = __builtin_amdgcn_wmma_f32_16x16x32_f16(false, a, false, bm2, (short)0, acc2, false, false);
      acc3 = __builtin_amdgcn_wmma_f32_16x16x32_f16(false, a, false, bm3, (short)0, acc3, false, false);
    }
    int kbase = mt * 16 + 8 * lg;
    int kb2 = kbase >> 5;
    int lsel = ((kbase & 31) >= 16) ? 16 : 0;
    int lslot = kbase & 15;
    v8f accs[4] = { acc0, acc1, acc2, acc3 };
    for (int nt = 0; nt < 4; nt++) {
      __align__(16) _Float16 hv[8];
#pragma unroll
      for (int v = 0; v < 8; v++) {
        float r = accs[nt][v] + bias[v];
        hv[v] = (_Float16)fmaxf(r, 0.0f);
      }
      *(float4*)(&A0[((nt * KB1 + kb2) * 32 + nl + lsel) * 16 + lslot]) = *(const float4*)hv;
    }
  }
  __syncthreads();

  // ---- layer 3: 1x256 dot + sigmoid + scatter ----
  if (tid < NPTS) {
    int n = tid, nt = n >> 4, nln = n & 15;
    float acc = b3g[0];
    for (int c = 0; c < 256; c++) {
      int kb = c >> 5, ki = c & 31;
      int lane3 = nln + ((ki >= 16) ? 16 : 0);
      float h = (float)A0[((nt * KB1 + kb) * 32 + lane3) * 16 + (ki & 15)];
      acc += w3s[c] * h;
    }
    float resv = 1.0f / (1.0f + __expf(-acc));
    if (pval[n]) out[b * HWSZ + pidx[n]] = resv;
  }
}

extern "C" void kernel_launch(void* const* d_in, const int* in_sizes, int n_in,
                              void* d_out, int out_size, void* d_ws, size_t ws_size,
                              hipStream_t stream) {
  const float* unc  = (const float*)d_in[0];
  const float* prev = (const float*)d_in[1];
  const float* enc  = (const float*)d_in[2];
  const float* w1   = (const float*)d_in[3];
  const float* b1   = (const float*)d_in[4];
  const float* g1   = (const float*)d_in[5];
  const float* be1  = (const float*)d_in[6];
  const float* m1   = (const float*)d_in[7];
  const float* v1   = (const float*)d_in[8];
  const float* w2   = (const float*)d_in[9];
  const float* b2   = (const float*)d_in[10];
  const float* g2   = (const float*)d_in[11];
  const float* be2  = (const float*)d_in[12];
  const float* m2   = (const float*)d_in[13];
  const float* v2   = (const float*)d_in[14];
  const float* w3   = (const float*)d_in[15];
  const float* b3   = (const float*)d_in[16];
  float* out = (float*)d_out;
  char* ws = (char*)d_ws;

  copy_mask_kernel<<<(BB * HWSZ / 4) / 256, 256, 0, stream>>>((const float4*)prev, (float4*)out);
  sel_init_kernel<<<1, 256, 0, stream>>>(ws);
  for (int r = 0; r < 4; r++) {
    sel_hist_kernel<<<256, 256, 0, stream>>>(unc, ws, r);
    sel_scan_kernel<<<1, 256, 0, stream>>>(ws);
  }
  sel_compact_kernel<<<256, 256, 0, stream>>>(unc, ws);
  prep_weights_kernel<<<546, 256, 0, stream>>>(w1, b1, g1, be1, m1, v1,
                                               w2, b2, g2, be2, m2, v2, ws);
  hier_main_kernel<<<BB * NTILES_PT, 256, 0, stream>>>(prev, enc, w3, b3, ws, out);
}
